// Attention_70111046140097
// MI455X (gfx1250) — matrix operand order repaired and checked by reading, not compiled, not run on hardware
//
#include <hip/hip_runtime.h>
#include <hip/hip_bf16.h>

typedef __attribute__((ext_vector_type(16))) _Float16 v16h;
typedef __attribute__((ext_vector_type(8)))  float    v8f;
typedef unsigned int v4u __attribute__((ext_vector_type(4)));
typedef int          v8i __attribute__((ext_vector_type(8)));
typedef int          v4i __attribute__((ext_vector_type(4)));

#define NUM_TOKENS 2048
#define NUM_HEADS  32
#define NUM_KVH    8
#define HEAD_DIM   128
#define QBLK       128   // query rows per workgroup (8 waves x 16 rows)
#define KBLK       32    // keys per inner iteration
// Fold softmax scale and log2(e) into Q so softmax runs in the exp2 domain.
#define QSCALE (0.08838834764831845f * 1.44269504088896340736f)

#if defined(__has_builtin)
#  if __has_builtin(__builtin_amdgcn_tensor_load_to_lds)
#    define HAVE_TDM 1
#  else
#    define HAVE_TDM 0
#  endif
#else
#  define HAVE_TDM 0
#endif

// Single-instruction max (skips fmaxf's sNaN canonicalization: v_max x,x).
__device__ __forceinline__ float fmax_raw(float a, float b) {
    float r;
    asm("v_max_num_f32 %0, %1, %2" : "=v"(r) : "v"(a), "v"(b));
    return r;
}

// Bare v_exp_f32 (hardware exp2): denormal results flush to zero, which is
// exactly right for softmax probabilities. Avoids libm's subnormal fixup
// (cndmask + ldexp) around every exp.
__device__ __forceinline__ float exp2_raw(float x) {
#if defined(__has_builtin) && __has_builtin(__builtin_amdgcn_exp2f)
    return __builtin_amdgcn_exp2f(x);
#else
    return exp2f(x);
#endif
}

// ---- ds_swizzle xor-lane exchange (immediate pattern, no index VALU) ----
template <int MASK>
__device__ __forceinline__ float swz_xor(float x) {
    int r = __builtin_amdgcn_ds_swizzle(__builtin_bit_cast(int, x),
                                        (MASK << 10) | 0x1f);  // group-of-32 xor
    return __builtin_bit_cast(float, r);
}
__device__ __forceinline__ float red_max16(float x) {
    x = fmax_raw(x, swz_xor<1>(x)); x = fmax_raw(x, swz_xor<2>(x));
    x = fmax_raw(x, swz_xor<4>(x)); x = fmax_raw(x, swz_xor<8>(x));
    return x;
}
__device__ __forceinline__ float red_sum16(float x) {
    x += swz_xor<1>(x); x += swz_xor<2>(x);
    x += swz_xor<4>(x); x += swz_xor<8>(x);
    return x;
}

#if HAVE_TDM
// ---- Tensor Data Mover: 2D tile of 16-bit data -> LDS ----
// rowElems/nRows/rowStrideElems in element (data_size=2B) units.
__device__ __forceinline__ void tdm_load_tile16(unsigned ldsOff, const void* gptr,
                                                unsigned rowElems, unsigned nRows,
                                                unsigned rowStrideElems) {
    unsigned long long ga = (unsigned long long)(uintptr_t)gptr;
    v4u g0 = { 1u,                                   // count=1, user descriptor
               ldsOff,                               // lds_addr (bytes)
               (unsigned)ga,                         // global_addr[31:0]
               (unsigned)((ga >> 32) & 0x01ffffffu) | (2u << 30) }; // addr hi | type=2
    v8i g1 = { (int)(1u << 16),                      // data_size = 2 bytes
               (int)(rowElems << 16),                // tensor_dim0[15:0]
               (int)(nRows << 16),                   // dim0 hi=0 | tensor_dim1[15:0]
               (int)(rowElems << 16),                // dim1 hi=0 | tile_dim0
               (int)nRows,                           // tile_dim1 | tile_dim2=0
               (int)rowStrideElems,                  // tensor_dim0_stride[31:0]
               0, 0 };
    v4i gz4 = { 0, 0, 0, 0 };
#if __clang_major__ >= 23
    v8i gz8 = { 0, 0, 0, 0, 0, 0, 0, 0 };
    __builtin_amdgcn_tensor_load_to_lds(g0, g1, gz4, gz4, gz8, 0);
#else
    __builtin_amdgcn_tensor_load_to_lds(g0, g1, gz4, gz4, 0);
#endif
}
#endif

// ---- Pre-pass: K -> f16 row-major; V -> f16 transposed (kvh, dim, token) ----
__global__ __launch_bounds__(256)
void convert_kv_kernel(const float* __restrict__ k, const float* __restrict__ v,
                       _Float16* __restrict__ k16, _Float16* __restrict__ vT) {
    const size_t i = (size_t)blockIdx.x * blockDim.x + threadIdx.x;
    k16[i] = (_Float16)k[i];
    const int d   = (int)(i % HEAD_DIM);
    const int kvh = (int)((i / HEAD_DIM) % NUM_KVH);
    const int t   = (int)(i / (HEAD_DIM * NUM_KVH));
    vT[((size_t)kvh * HEAD_DIM + d) * NUM_TOKENS + t] = (_Float16)v[i];
}

__global__ __launch_bounds__(256, 1)
void fa_prefill_kernel(const float* __restrict__ q,
                       const float* __restrict__ k,
                       const float* __restrict__ v,
                       const _Float16* __restrict__ k16,
                       const _Float16* __restrict__ vT,
                       float* __restrict__ out,
                       int useTDM)
{
    __shared__ _Float16 Ks16[2][KBLK][HEAD_DIM];  // f16 K tile (key, dim), double-buffered
    __shared__ _Float16 Vt16[2][HEAD_DIM][KBLK];  // f16 V tile (dim, key), double-buffered
    __shared__ _Float16 Ps[8][16][KBLK];          // per-wave P tile (C->A relayout)

    const int qb   = blockIdx.x;
    const int h    = blockIdx.y;
    const int kvh  = h / (NUM_HEADS / NUM_KVH);
    const int tid  = threadIdx.x;
    const int wave = tid >> 5;
    const int lane = tid & 31;
    const int l16  = lane & 15;
    const bool hi  = lane >= 16;

#if HAVE_TDM
    const bool tdmPath = (useTDM != 0);
#else
    const bool tdmPath = false; (void)useTDM; (void)k16; (void)vT;
#endif

    const int qRow0  = qb * QBLK + wave * 16;
    const int myQRow = qRow0 + l16;

    // ---- Preload Q A-fragments: f32 -> f16, then packed-f16 scale ----
    v16h qf[4];
    {
        const float* qp = q + ((size_t)myQRow * NUM_HEADS + h) * HEAD_DIM;
        const _Float16 qs = (_Float16)QSCALE;
        #pragma unroll
        for (int kk = 0; kk < 4; ++kk) {
            const int off = kk * 32 + (hi ? 16 : 0);
            #pragma unroll
            for (int i = 0; i < 8; ++i) {
                qf[kk][i]     = (_Float16)qp[off + i];
                qf[kk][8 + i] = (_Float16)qp[off + 8 + i];
            }
            qf[kk] = qf[kk] * qs;   // v_pk_mul_f16
        }
    }

    v8f o[8];
    float m[8], l[8];
    {
        v8f zero = {};
        #pragma unroll
        for (int n = 0; n < 8; ++n) o[n] = zero;
        #pragma unroll
        for (int r = 0; r < 8; ++r) { m[r] = -INFINITY; l[r] = 0.0f; }
    }

    const int jEnd = (qb + 1) * (QBLK / KBLK);

#if HAVE_TDM
    if (tdmPath && wave == 0) {      // kick off tile 0 DMA
        tdm_load_tile16((unsigned)(uintptr_t)&Ks16[0][0][0],
                        k16 + (size_t)kvh * HEAD_DIM,
                        HEAD_DIM, KBLK, NUM_KVH * HEAD_DIM);
        tdm_load_tile16((unsigned)(uintptr_t)&Vt16[0][0][0],
                        vT + (size_t)kvh * HEAD_DIM * NUM_TOKENS,
                        KBLK, HEAD_DIM, NUM_TOKENS);
    }
#endif

    for (int j = 0; j < jEnd; ++j) {
        const int kBase = j * KBLK;
        const int buf   = j & 1;

        if (tdmPath) {
#if HAVE_TDM
            if (wave == 0) __builtin_amdgcn_s_wait_tensorcnt(0);
            __syncthreads();   // tile j visible; everyone done with tile j-1
            if (wave == 0 && (j + 1) < jEnd) {  // prefetch tile j+1
                const int nb = kBase + KBLK;
                tdm_load_tile16((unsigned)(uintptr_t)&Ks16[buf ^ 1][0][0],
                                k16 + ((size_t)nb * NUM_KVH + kvh) * HEAD_DIM,
                                HEAD_DIM, KBLK, NUM_KVH * HEAD_DIM);
                tdm_load_tile16((unsigned)(uintptr_t)&Vt16[buf ^ 1][0][0],
                                vT + (size_t)kvh * HEAD_DIM * NUM_TOKENS + nb,
                                KBLK, HEAD_DIM, NUM_TOKENS);
            }
#endif
        } else {
            __syncthreads();   // protect tiles from previous iteration's readers
            const int key = tid >> 3;
            const int ch  = tid & 7;
            const float* kp = k + ((size_t)(kBase + key) * NUM_KVH + kvh) * HEAD_DIM + ch * 16;
            const float* vp = v + ((size_t)(kBase + key) * NUM_KVH + kvh) * HEAD_DIM + ch * 16;
            #pragma unroll
            for (int i = 0; i < 16; ++i) {
                Ks16[buf][key][ch * 16 + i] = (_Float16)kp[i];
                Vt16[buf][ch * 16 + i][key] = (_Float16)vp[i];
            }
            __syncthreads();
        }

        // Skip tiles fully above this wave's rows (wave-uniform)
        if (kBase <= qRow0 + 15) {
            // ---- S = (Q*scale) * K^T : pure ds_load_b128 -> WMMA ----
            v8f s0 = {}, s1 = {};
            #pragma unroll
            for (int kk = 0; kk < 4; ++kk) {
                const v16h b0 = *reinterpret_cast<const v16h*>(
                    &Ks16[buf][l16     ][kk * 32 + (hi ? 16 : 0)]);
                const v16h b1 = *reinterpret_cast<const v16h*>(
                    &Ks16[buf][l16 + 16][kk * 32 + (hi ? 16 : 0)]);
                s0 = __builtin_amdgcn_wmma_f32_16x16x32_f16(false, qf[kk], false, b0,
                                                            (short)0, s0, false, false);
                s1 = __builtin_amdgcn_wmma_f32_16x16x32_f16(false, qf[kk], false, b1,
                                                            (short)0, s1, false, false);
            }

            // ---- Causal mask: only diagonal tiles, isolated from softmax ----
            const bool needMask = (kBase + KBLK - 1) > qRow0;  // wave-uniform
            if (needMask) {
                #pragma unroll
                for (int r = 0; r < 8; ++r) {
                    const int row = qRow0 + r + (hi ? 8 : 0);
                    if (kBase + l16      > row) s0[r] = -1.0e30f;
                    if (kBase + l16 + 16 > row) s1[r] = -1.0e30f;
                }
            }

            // ---- Online softmax (exp2 domain) ----
            v8f corrv;
            #pragma unroll
            for (int r = 0; r < 8; ++r) {
                const float mx   = red_max16(fmax_raw(s0[r], s1[r]));
                const float mnew = fmax_raw(m[r], mx);
                const float p0   = exp2_raw(s0[r] - mnew);
                const float p1   = exp2_raw(s1[r] - mnew);
                const float sum  = red_sum16(p0 + p1);
                const float corr = exp2_raw(m[r] - mnew);
                m[r] = mnew;
                l[r] = l[r] * corr + sum;
                corrv[r] = corr;

                Ps[wave][r + (hi ? 8 : 0)][l16]      = (_Float16)p0;
                Ps[wave][r + (hi ? 8 : 0)][l16 + 16] = (_Float16)p1;
            }
            // Rescale accumulators with vector muls (pk/dual-issue friendly)
            #pragma unroll
            for (int n = 0; n < 8; ++n) o[n] = o[n] * corrv;

            // ---- Reload P as 16x32 A-fragment (same wave; DS in-order) ----
            v16h pf;
            {
                const _Float16* pp = &Ps[wave][l16][hi ? 16 : 0];
                #pragma unroll
                for (int i = 0; i < 8; ++i) { pf[i] = pp[i]; pf[8 + i] = pp[8 + i]; }
            }

            // ---- O += P * V ----
            #pragma unroll
            for (int n = 0; n < 8; ++n) {
                const v16h vb = *reinterpret_cast<const v16h*>(
                    &Vt16[buf][n * 16 + l16][hi ? 16 : 0]);
                o[n] = __builtin_amdgcn_wmma_f32_16x16x32_f16(false, pf, false, vb,
                                                              (short)0, o[n], false, false);
            }
        }
    }

    // ---- Normalize and write out ----
    #pragma unroll
    for (int r = 0; r < 8; ++r) {
        const float inv = 1.0f / l[r];
        const int row = qRow0 + r + (hi ? 8 : 0);
        float* op = out + ((size_t)row * NUM_HEADS + h) * HEAD_DIM;
        #pragma unroll
        for (int n = 0; n < 8; ++n) op[n * 16 + l16] = o[n][r] * inv;
    }
}

extern "C" void kernel_launch(void* const* d_in, const int* in_sizes, int n_in,
                              void* d_out, int out_size, void* d_ws, size_t ws_size,
                              hipStream_t stream) {
    (void)in_sizes; (void)n_in; (void)out_size;
    const float* q = (const float*)d_in[0];
    const float* k = (const float*)d_in[1];
    const float* v = (const float*)d_in[2];
    float* out = (float*)d_out;

    const size_t kvElems  = (size_t)NUM_TOKENS * NUM_KVH * HEAD_DIM;  // 2M
    const size_t kvBytes  = kvElems * sizeof(_Float16);               // 4 MB
    const size_t wsNeeded = 2 * kvBytes;                              // 8 MB
    const int useTDM = (d_ws != nullptr && ws_size >= wsNeeded) ? 1 : 0;
    _Float16* k16 = (_Float16*)d_ws;
    _Float16* vT  = (_Float16*)((char*)d_ws + kvBytes);

    if (useTDM) {
        convert_kv_kernel<<<(unsigned)(kvElems / 256), 256, 0, stream>>>(k, v, k16, vT);
    }

    dim3 grid(NUM_TOKENS / QBLK, NUM_HEADS);   // (16, 32)
    dim3 block(256);                           // 8 waves of 32
    fa_prefill_kernel<<<grid, block, 0, stream>>>(q, k, v, k16, vT, out, useTDM);
}
